// Decoder_17841294148315
// MI455X (gfx1250) — matrix operand order repaired and checked by reading, not compiled, not run on hardware
//
#include <hip/hip_runtime.h>
#include <hip/hip_bf16.h>
#include <math.h>

// ---------------------------------------------------------------------------
// Decoder: X_hat.T = gen_Z @ softmax(query@key^T/sqrt(32) + gumbel)^T
// Dominant cost: 16384x16384x512 NT-GEMM (275 GFLOP) + 1 GiB fp32 output.
// MI455X: use v_wmma_f32_16x16x32_f16 (wave32), LDS-staged tiles, f32 accum.
// ---------------------------------------------------------------------------

typedef __attribute__((ext_vector_type(16))) _Float16 v16h;
typedef __attribute__((ext_vector_type(8)))  float    v8f;

union Frag16 { v16h v; uint4 u[2]; };

#define ZC 16384      // cells
#define DZ 512        // z_dim
#define NG 16384      // genes
#define RG 128
#define KD 32
#define HH 256

__device__ __forceinline__ float gelu_exact(float x) {
    return 0.5f * x * (1.0f + erff(x * 0.70710678118654752f));
}

__device__ __forceinline__ unsigned rotl32(unsigned x, int n) {
    return (x << n) | (x >> (32 - n));
}

// threefry2x32-20, key = jax.random.key(42) -> (0, 42); JAX counter split:
// counts = iota(n); x0 = counts[:n/2], x1 = counts[n/2:]
__device__ __forceinline__ float gumbel_noise(unsigned idx) {
    const unsigned k0 = 0u, k1 = 42u;
    const unsigned ks = k0 ^ k1 ^ 0x1BD11BDAu;
    const unsigned halfn = (NG * DZ) / 2;   // 4194304
    const bool lo = idx < halfn;
    unsigned x0 = lo ? idx : idx - halfn;
    unsigned x1 = lo ? idx + halfn : idx;
    unsigned X0 = x0 + k0, X1 = x1 + k1;
#define TFR(r) { X0 += X1; X1 = rotl32(X1, (r)); X1 ^= X0; }
    TFR(13) TFR(15) TFR(26) TFR(6)  X0 += k1; X1 += ks + 1u;
    TFR(17) TFR(29) TFR(16) TFR(24) X0 += ks; X1 += k0 + 2u;
    TFR(13) TFR(15) TFR(26) TFR(6)  X0 += k0; X1 += k1 + 3u;
    TFR(17) TFR(29) TFR(16) TFR(24) X0 += k1; X1 += ks + 4u;
    TFR(13) TFR(15) TFR(26) TFR(6)  X0 += ks; X1 += k0 + 5u;
#undef TFR
    unsigned bits = lo ? X0 : X1;
    unsigned fb = (bits >> 9) | 0x3F800000u;   // [1,2)
    float u;
    __builtin_memcpy(&u, &fb, 4);
    u -= 1.0f;
    u = fmaxf(u, 1.1754944e-38f);              // uniform(minval=tiny)
    return -logf(-logf(u));
}

// --------------------------- K0: zero scratch ------------------------------
__global__ void k0_zero(float* __restrict__ p, int n) {
    int i = blockIdx.x * 256 + threadIdx.x;
    if (i < n) p[i] = 0.0f;
}

// ------------- K1: H1pre[d][h] += sum_c rawZ[c][d] * W1z[c][h] -------------
// grid (32 d-tiles, 16 c-splits), block 256 (one h each), split-K atomics.
__global__ __launch_bounds__(256) void k1_hidden(const float* __restrict__ rawZ,
                                                 const float* __restrict__ W1z,
                                                 float* __restrict__ H1pre) {
    __shared__ float za[128 * 16];
    const int h = threadIdx.x;
    const int d0 = blockIdx.x * 16;
    const int cbase = blockIdx.y * 1024;
    float acc[16];
#pragma unroll
    for (int di = 0; di < 16; ++di) acc[di] = 0.0f;

    for (int cc = 0; cc < 1024; cc += 128) {
        // stage rawZ[cbase+cc .. +128)[d0 .. d0+16) -> LDS (2 x float4/thread)
        const int r = h >> 1, h8 = (h & 1) * 8;
        const float* src = rawZ + (size_t)(cbase + cc + r) * DZ + d0 + h8;
        float4 v0 = *(const float4*)(src);
        float4 v1 = *(const float4*)(src + 4);
        *(float4*)(za + r * 16 + h8)     = v0;
        *(float4*)(za + r * 16 + h8 + 4) = v1;
        __syncthreads();
        for (int c = 0; c < 128; ++c) {
            const float w = W1z[(size_t)(cbase + cc + c) * HH + h];
#pragma unroll
            for (int di = 0; di < 16; ++di) acc[di] += za[c * 16 + di] * w;
        }
        __syncthreads();
    }
#pragma unroll
    for (int di = 0; di < 16; ++di)
        atomicAdd(&H1pre[(size_t)(d0 + di) * HH + h], acc[di]);
}

// -------------------- K1b: H1 = gelu(H1pre + b1z) (in place) ---------------
__global__ void k1b_gelu(float* __restrict__ H1, const float* __restrict__ b1z) {
    int i = blockIdx.x * 256 + threadIdx.x;     // i < 512*256
    int h = i & (HH - 1);
    H1[i] = gelu_exact(H1[i] + b1z[h]);
}

// ------------------- K1c: key[d][k] = H1[d] @ W2z[:,k] + b2z ---------------
__global__ void k1c_key(const float* __restrict__ H1, const float* __restrict__ W2z,
                        const float* __restrict__ b2z, float* __restrict__ keyv) {
    const int d = blockIdx.x, k = threadIdx.x;  // block 32
    float s = b2z[k];
    for (int h = 0; h < HH; ++h) s += H1[(size_t)d * HH + h] * W2z[h * KD + k];
    keyv[(size_t)d * KD + k] = s;
}

// ---------------- K2: query = gelu(G_rep@W1g+b1g)@W2g + b2g ----------------
// 4 genes per 128-thread block, one wave32 per gene.
__global__ __launch_bounds__(128) void k2_query(const float* __restrict__ Grep,
                                                const float* __restrict__ W1g,
                                                const float* __restrict__ b1g,
                                                const float* __restrict__ W2g,
                                                const float* __restrict__ b2g,
                                                float* __restrict__ query) {
    __shared__ float grow[4][RG];
    __shared__ float hid[4][KD];
    const int sub = threadIdx.x >> 5, lane = threadIdx.x & 31;
    const size_t g = (size_t)blockIdx.x * 4 + sub;
#pragma unroll
    for (int j = 0; j < 4; ++j)
        grow[sub][lane + j * 32] = Grep[g * RG + lane + j * 32];
    __syncthreads();
    float s = b1g[lane];
    for (int c = 0; c < RG; ++c) s += grow[sub][c] * W1g[c * KD + lane];
    hid[sub][lane] = gelu_exact(s);
    __syncthreads();
    float qv = b2g[lane];
#pragma unroll
    for (int h = 0; h < KD; ++h) qv += hid[sub][h] * W2g[h * KD + lane];
    query[g * KD + lane] = qv;
}

// ------- K3: scores + gumbel + softmax(axis=d) -> p_attn (f16) -------------
// one block (512 threads = 16 waves) per gene row.
__global__ __launch_bounds__(512) void k3_softmax(const float* __restrict__ query,
                                                  const float* __restrict__ keyv,
                                                  _Float16* __restrict__ pah) {
    __shared__ float q[KD];
    __shared__ float red[DZ];
    const int d = threadIdx.x;
    const unsigned g = blockIdx.x;
    if (d < KD) q[d] = query[(size_t)g * KD + d];
    __syncthreads();
    float s = 0.0f;
    const float* kr = keyv + (size_t)d * KD;
#pragma unroll
    for (int k = 0; k < KD; ++k) s += q[k] * kr[k];
    s *= 0.17677669529663687f;                 // 1/sqrt(32)
    s += gumbel_noise(g * DZ + d);             // tau = 1
    red[d] = s;
    __syncthreads();
    for (int o = 256; o > 0; o >>= 1) {
        if (d < o) red[d] = fmaxf(red[d], red[d + o]);
        __syncthreads();
    }
    const float mx = red[0];
    __syncthreads();
    const float e = expf(s - mx);
    red[d] = e;
    __syncthreads();
    for (int o = 256; o > 0; o >>= 1) {
        if (d < o) red[d] += red[d + o];
        __syncthreads();
    }
    pah[(size_t)g * DZ + d] = (_Float16)(e / red[0]);
}

// ----------------------- K4: gen_Z fp32 -> f16 -----------------------------
__global__ void k4_cvt(const float* __restrict__ src, _Float16* __restrict__ dst) {
    const int i = (blockIdx.x * 256 + threadIdx.x) * 4;
    float4 v = *(const float4*)(src + i);
    union { _Float16 h[4]; uint2 u; } o;
    o.h[0] = (_Float16)v.x; o.h[1] = (_Float16)v.y;
    o.h[2] = (_Float16)v.z; o.h[3] = (_Float16)v.w;
    *(uint2*)(dst + i) = o.u;
}

// ------------ K5: out[c][g] = sum_d genZ_h[c][d] * p_attn_h[g][d] ----------
// NT GEMM, 128x128 tile, BK=32, 8 wave32 waves (2x4), each wave 64x32,
// v_wmma_f32_16x16x32_f16, double-buffered LDS (80 B padded rows -> no bank
// conflicts, 16B-aligned ds_load_b128 fragment reads).
#define LDSTRIDE 80
#define TILEBYTES (128 * LDSTRIDE)           // 10240 per matrix
#define BUFBYTES  (2 * TILEBYTES)            // 20480 per stage

__device__ __forceinline__ void load_tile_regs(const _Float16* __restrict__ gA,
                                               const _Float16* __restrict__ gB,
                                               int k0, int tid,
                                               uint4& a0, uint4& a1, uint4& b0, uint4& b1) {
    const int r = tid >> 2, cs = (tid & 3) * 8;
    a0 = *(const uint4*)(gA + (size_t)r        * DZ + k0 + cs);
    a1 = *(const uint4*)(gA + (size_t)(r + 64) * DZ + k0 + cs);
    b0 = *(const uint4*)(gB + (size_t)r        * DZ + k0 + cs);
    b1 = *(const uint4*)(gB + (size_t)(r + 64) * DZ + k0 + cs);
}

__device__ __forceinline__ void store_tile_lds(unsigned char* bufA, unsigned char* bufB,
                                               int tid, uint4 a0, uint4 a1, uint4 b0, uint4 b1) {
    const int r = tid >> 2, cs = (tid & 3) * 16;
    *(uint4*)(bufA + r        * LDSTRIDE + cs) = a0;
    *(uint4*)(bufA + (r + 64) * LDSTRIDE + cs) = a1;
    *(uint4*)(bufB + r        * LDSTRIDE + cs) = b0;
    *(uint4*)(bufB + (r + 64) * LDSTRIDE + cs) = b1;
}

__global__ __launch_bounds__(256) void k5_gemm(const _Float16* __restrict__ Ah,  // genZ f16 [16384][512]
                                               const _Float16* __restrict__ Bh,  // p_attn f16 [16384][512]
                                               float* __restrict__ C) {          // [16384 cells][16384 genes]
    __shared__ __align__(16) unsigned char smem[2 * BUFBYTES];   // 40 KB
    const int tid  = threadIdx.x;
    const int lane = tid & 31, wave = tid >> 5;
    const int wm = wave & 1, wn = wave >> 1;        // 2 x 4 wave grid
    const int sel = lane >> 4, mr = lane & 15;

    const size_t cblk = (size_t)blockIdx.y * 128;
    const size_t gblk = (size_t)blockIdx.x * 128;
    const _Float16* gA = Ah + cblk * DZ;
    const _Float16* gB = Bh + gblk * DZ;

    v8f acc[4][2] = {};

    uint4 a0, a1, b0, b1;
    load_tile_regs(gA, gB, 0, tid, a0, a1, b0, b1);
    store_tile_lds(smem, smem + TILEBYTES, tid, a0, a1, b0, b1);
    __syncthreads();

    for (int kk = 0; kk < DZ / 32; ++kk) {
        const int cur = kk & 1;
        if (kk < DZ / 32 - 1)
            load_tile_regs(gA, gB, (kk + 1) * 32, tid, a0, a1, b0, b1);

        unsigned char* bufA = smem + cur * BUFBYTES;
        unsigned char* bufB = bufA + TILEBYTES;

        // B fragments: per-lane row = N, K 0..7,16..23 (or 8..15,24..31)
        Frag16 fb[2];
#pragma unroll
        for (int ni = 0; ni < 2; ++ni) {
            const unsigned char* p = bufB + (wn * 32 + ni * 16 + mr) * LDSTRIDE + sel * 16;
            fb[ni].u[0] = *(const uint4*)p;
            fb[ni].u[1] = *(const uint4*)(p + 32);
        }
#pragma unroll
        for (int mi = 0; mi < 4; ++mi) {
            Frag16 fa;
            const unsigned char* p = bufA + (wm * 64 + mi * 16 + mr) * LDSTRIDE + sel * 16;
            fa.u[0] = *(const uint4*)p;          // K 0..7  (or 8..15)
            fa.u[1] = *(const uint4*)(p + 32);   // K 16..23 (or 24..31)
#pragma unroll
            for (int ni = 0; ni < 2; ++ni) {
                acc[mi][ni] = __builtin_amdgcn_wmma_f32_16x16x32_f16(
                    false, fa.v, false, fb[ni].v, (short)0, acc[mi][ni], false, false);
            }
        }

        if (kk < DZ / 32 - 1) {
            unsigned char* nA = smem + (cur ^ 1) * BUFBYTES;
            store_tile_lds(nA, nA + TILEBYTES, tid, a0, a1, b0, b1);
            __syncthreads();
        }
    }

    // epilogue: C/D layout -> lanes 0-15: (M=r, N=lane); 16-31: (M=r+8, N=lane-16)
#pragma unroll
    for (int mi = 0; mi < 4; ++mi) {
#pragma unroll
        for (int ni = 0; ni < 2; ++ni) {
            const size_t crow = cblk + wm * 64 + mi * 16 + sel * 8;
            const size_t gcol = gblk + wn * 32 + ni * 16 + mr;
            float* cp = C + crow * NG + gcol;
#pragma unroll
            for (int r = 0; r < 8; ++r)
                cp[(size_t)r * NG] = acc[mi][ni][r];
        }
    }
}

// ---------------------------------------------------------------------------
extern "C" void kernel_launch(void* const* d_in, const int* in_sizes, int n_in,
                              void* d_out, int out_size, void* d_ws, size_t ws_size,
                              hipStream_t stream) {
    const float* rawZ = (const float*)d_in[0];
    const float* genZ = (const float*)d_in[1];
    const float* W1z  = (const float*)d_in[2];
    const float* b1z  = (const float*)d_in[3];
    const float* W2z  = (const float*)d_in[4];
    const float* b2z  = (const float*)d_in[5];
    const float* W1g  = (const float*)d_in[6];
    const float* b1g  = (const float*)d_in[7];
    const float* W2g  = (const float*)d_in[8];
    const float* b2g  = (const float*)d_in[9];
    const float* Grep = (const float*)d_in[10];
    float* out = (float*)d_out;

    // workspace carve (~36 MB)
    float* H1   = (float*)d_ws;                         // 512*256 f32 (pre + post gelu, in place)
    float* keyv = H1 + (size_t)DZ * HH;                 // 512*32
    float* query = keyv + (size_t)DZ * KD;              // 16384*32
    _Float16* pah = (_Float16*)(query + (size_t)NG * KD);   // 16384*512 f16
    _Float16* gzh = pah + (size_t)NG * DZ;                  // 16384*512 f16

    k0_zero<<<(DZ * HH + 255) / 256, 256, 0, stream>>>(H1, DZ * HH);
    k1_hidden<<<dim3(DZ / 16, 16), 256, 0, stream>>>(rawZ, W1z, H1);
    k1b_gelu<<<(DZ * HH) / 256, 256, 0, stream>>>(H1, b1z);
    k1c_key<<<DZ, KD, 0, stream>>>(H1, W2z, b2z, keyv);
    k2_query<<<NG / 4, 128, 0, stream>>>(Grep, W1g, b1g, W2g, b2g, query);
    k3_softmax<<<NG, DZ, 0, stream>>>(query, keyv, pah);
    k4_cvt<<<(ZC * DZ) / 1024, 256, 0, stream>>>(genZ, gzh);
    k5_gemm<<<dim3(NG / 128, ZC / 128), 256, 0, stream>>>(gzh, pah, out);
}
